// MTRNN_9929964389009
// MI455X (gfx1250) — compile-verified, matrix operand order checked by reading, not run on hardware
//
#include <hip/hip_runtime.h>
#include <math.h>

// ---------------------------------------------------------------------------
// MTRNN step, dead-code-eliminated to the live dataflow:
//   io_new = tanh(0.5*io_state + 0.5*([io|cf|x] @ [Wio2io|Wcf2io|Wi2io]^T + bsum))
//   y      = tanh(io_new @ Wio2o^T + bio2o)
// Memory-bound (~43 MB -> ~1.8 us at 23.3 TB/s); bf16 WMMA w/ f32 accum.
// Stage-1 weights are staged per-block into LDS in fragment order (double-
// buffered), so the hot loop is ds_load_b128 x2 -> v_wmma with no hazard nops.
// ---------------------------------------------------------------------------

typedef __attribute__((ext_vector_type(16))) __bf16 v16bf;
typedef __attribute__((ext_vector_type(8)))  float  v8f;

#define BATCH   8192
#define IO_DIM  512
#define IN_DIM  128
#define ROWS_PER_BLOCK 64
#define NKC 36          // K = 512 + 512 + 128 = 1152 -> 36 chunks of 32

__device__ __forceinline__ float fast_tanh(float v) {
#if __has_builtin(__builtin_amdgcn_tanhf)
  return __builtin_amdgcn_tanhf(v);
#else
  return tanhf(v);
#endif
}

// Pack 16 fp32 into a bf16 fragment (element order = CDNA5 16-bit A/B layout).
__device__ __forceinline__ v16bf pack_frag(float4 a, float4 b, float4 c, float4 d) {
  v16bf r;
  r[0]  = (__bf16)a.x; r[1]  = (__bf16)a.y; r[2]  = (__bf16)a.z; r[3]  = (__bf16)a.w;
  r[4]  = (__bf16)b.x; r[5]  = (__bf16)b.y; r[6]  = (__bf16)b.z; r[7]  = (__bf16)b.w;
  r[8]  = (__bf16)c.x; r[9]  = (__bf16)c.y; r[10] = (__bf16)c.z; r[11] = (__bf16)c.w;
  r[12] = (__bf16)d.x; r[13] = (__bf16)d.y; r[14] = (__bf16)d.z; r[15] = (__bf16)d.w;
  return r;
}

// Gather a lane's fragment straight from fp32 memory: 8 floats at p, 8 at p+16.
__device__ __forceinline__ v16bf frag_from_f32(const float* __restrict__ p) {
  const float4* q = (const float4*)p;
  return pack_frag(q[0], q[1], q[4], q[5]);
}

__global__ __launch_bounds__(256)
void mtrnn_fused(const float* __restrict__ x,
                 const float* __restrict__ io_state,
                 const float* __restrict__ cf_state,
                 const float* __restrict__ Wi2io,  const float* __restrict__ bi2io,
                 const float* __restrict__ Wio2o,  const float* __restrict__ bio2o,
                 const float* __restrict__ Wio2io, const float* __restrict__ bio2io,
                 const float* __restrict__ Wcf2io, const float* __restrict__ bcf2io,
                 float* __restrict__ out)
{
  // 64 KB LDS, time-multiplexed:
  //   stage 1 k-loop : wbuf0/wbuf1 = 2 x 32 KB weight chunks (fragment order)
  //   stage 2        : abuf        = 64 KB io_new in A-fragment order
  __shared__ __align__(64) char smem_raw[65536];
  __bf16* wbuf0 = (__bf16*)&smem_raw[0];
  __bf16* wbuf1 = (__bf16*)&smem_raw[32768];
  __bf16* abuf  = (__bf16*)&smem_raw[0];

  const int tid      = threadIdx.x;
  const int wave     = tid >> 5;        // 8 wave32 waves
  const int lane     = tid & 31;
  const int rowGroup = wave >> 1;       // 0..3 : 16-row slab
  const int nHalf    = wave & 1;        // 0..1 : 256-col half (stage 1)
  const int col      = lane & 15;
  const int khi      = lane >> 4;       // K sub-half per 16-bit fragment layout
  const int klo      = khi * 8;
  const int rOff     = khi * 8;         // C-tile row offset: M = v + 8*khi

  const int waveRow  = blockIdx.x * ROWS_PER_BLOCK + rowGroup * 16;
  const int aRow     = waveRow + col;

  // -- per-kc source selection ---------------------------------------------
  auto sel = [&](int kc, const float*& Ab, const float*& Wb, int& lda, int& koff) {
    if (kc < 16)      { Ab = io_state; Wb = Wio2io; lda = 512; koff = kc * 32; }
    else if (kc < 32) { Ab = cf_state; Wb = Wcf2io; lda = 512; koff = (kc - 16) * 32; }
    else              { Ab = x;        Wb = Wi2io;  lda = 128; koff = (kc - 32) * 32; }
  };

  // -- cooperative weight staging: 512 rows x 32 k, 2 rows per thread ------
  float4 g[2][8];
  auto issue = [&](int kc) {            // kick off global loads for chunk kc
    const float *Ab, *Wb; int lda, koff; sel(kc, Ab, Wb, lda, koff);
    #pragma unroll
    for (int rep = 0; rep < 2; ++rep) {
      const float4* src = (const float4*)(Wb + (size_t)(tid + rep * 256) * lda + koff);
      #pragma unroll
      for (int i = 0; i < 8; ++i) g[rep][i] = src[i];
    }
  };
  auto flush = [&](__bf16* buf) {       // cvt + store in fragment slot order
    #pragma unroll
    for (int rep = 0; rep < 2; ++rep) {
      __bf16* dst = buf + (tid + rep * 256) * 32;
      // slot h=0: k {0..7,16..23}   slot h=1: k {8..15,24..31}
      *(v16bf*)(dst)      = pack_frag(g[rep][0], g[rep][1], g[rep][4], g[rep][5]);
      *(v16bf*)(dst + 16) = pack_frag(g[rep][2], g[rep][3], g[rep][6], g[rep][7]);
    }
  };

  // ---------------- stage 1: pre = [io|cf|x] @ Wcat^T ----------------------
  v8f acc[16];
  #pragma unroll
  for (int i = 0; i < 16; ++i) acc[i] = (v8f){0.f,0.f,0.f,0.f,0.f,0.f,0.f,0.f};

  issue(0);
  flush(wbuf0);
  __syncthreads();

  for (int kc = 0; kc < NKC; ++kc) {
    const float *Ab, *Wb; int lda, koff; sel(kc, Ab, Wb, lda, koff);
    if (kc + 1 < NKC) issue(kc + 1);    // overlap next chunk's loads w/ wmmas

    v16bf a = frag_from_f32(Ab + (size_t)aRow * lda + koff + klo);

    const __bf16* cur = (kc & 1) ? wbuf1 : wbuf0;
    const __bf16* bbase = cur + (nHalf * 256 + col) * 32 + khi * 16;
    #pragma unroll
    for (int nt = 0; nt < 16; ++nt) {
      v16bf b = *(const v16bf*)(bbase + nt * 512);   // 2x ds_load_b128
      acc[nt] = __builtin_amdgcn_wmma_f32_16x16x32_bf16(
          false, a, false, b, (short)0, acc[nt], false, false);
    }

    if (kc + 1 < NKC) flush((kc & 1) ? wbuf0 : wbuf1);
    __syncthreads();
  }
  // (final barrier above guarantees all wbuf reads retired -> abuf may alias)

  // ---- stage-1 epilogue: leak + bias + tanh, io_new -> LDS (frag order) ---
  #pragma unroll
  for (int nt = 0; nt < 16; ++nt) {
    const int n   = nHalf * 256 + nt * 16 + col;     // stage-2 K index
    const float bsum = bio2io[n] + bcf2io[n] + bi2io[n];
    const int r5  = n & 31;
    const int h   = (r5 >> 3) & 1;
    const int p   = (r5 & 7) | (((r5 >> 4) & 1) << 3);
    const int kc2 = n >> 5;
    #pragma unroll
    for (int v = 0; v < 8; ++v) {
      const int rl   = rowGroup * 16 + rOff + v;     // block-local row
      const int grow = waveRow + rOff + v;
      const float pre    = acc[nt][v] + bsum;
      const float io_old = io_state[(size_t)grow * IO_DIM + n];
      const float t      = fast_tanh(0.5f * io_old + 0.5f * pre);  // TAU_IO=2
      abuf[((rl * 16 + kc2) * 2 + h) * 16 + p] = (__bf16)t;
    }
  }
  __syncthreads();

  // ---------------- stage 2: y = tanh(io_new @ Wio2o^T + bio2o) ------------
  v8f acc2[4];
  #pragma unroll
  for (int i = 0; i < 4; ++i) acc2[i] = (v8f){0.f,0.f,0.f,0.f,0.f,0.f,0.f,0.f};

  const __bf16* a2base = abuf + ((rowGroup * 16 + col) * 16 * 2 + khi) * 16;

  for (int kc2 = 0; kc2 < 16; ++kc2) {               // K = 512
    v16bf a2 = *(const v16bf*)(a2base + kc2 * 32);   // contiguous 32B LDS read
    #pragma unroll
    for (int nt = 0; nt < 4; ++nt) {
      const int n = nHalf * 64 + nt * 16 + col;      // output col 0..127
      v16bf b = frag_from_f32(Wio2o + (size_t)n * IO_DIM + kc2 * 32 + klo);
      acc2[nt] = __builtin_amdgcn_wmma_f32_16x16x32_bf16(
          false, a2, false, b, (short)0, acc2[nt], false, false);
    }
  }

  #pragma unroll
  for (int nt = 0; nt < 4; ++nt) {
    const int n = nHalf * 64 + nt * 16 + col;
    const float bb = bio2o[n];
    #pragma unroll
    for (int v = 0; v < 8; ++v) {
      const int grow = waveRow + rOff + v;
      out[(size_t)grow * IN_DIM + n] = fast_tanh(acc2[nt][v] + bb);
    }
  }
}

// ---------------------------------------------------------------------------

extern "C" void kernel_launch(void* const* d_in, const int* in_sizes, int n_in,
                              void* d_out, int out_size, void* d_ws, size_t ws_size,
                              hipStream_t stream) {
  (void)in_sizes; (void)n_in; (void)d_ws; (void)ws_size; (void)out_size;
  // setup_inputs() dict order:
  // 0:x 1:io_state 2:cf_state 3:cs_state 4:Wi2io 5:bi2io 6:Wio2o 7:bio2o
  // 8:Wio2io 9:bio2io 10:Wio2cf 11:bio2cf 12:Wcf2io 13:bcf2io 14:Wcf2cs
  // 15:bcf2cs 16:Wcf2cf 17:bcf2cf 18:Wcs2cf 19:bcs2cf 20:Wcs2cs 21:bcs2cs
  const float* x        = (const float*)d_in[0];
  const float* io_state = (const float*)d_in[1];
  const float* cf_state = (const float*)d_in[2];
  const float* Wi2io    = (const float*)d_in[4];
  const float* bi2io    = (const float*)d_in[5];
  const float* Wio2o    = (const float*)d_in[6];
  const float* bio2o    = (const float*)d_in[7];
  const float* Wio2io   = (const float*)d_in[8];
  const float* bio2io   = (const float*)d_in[9];
  const float* Wcf2io   = (const float*)d_in[12];
  const float* bcf2io   = (const float*)d_in[13];
  float* out = (float*)d_out;

  dim3 grid(BATCH / ROWS_PER_BLOCK);  // 128 blocks
  dim3 block(256);                    // 8 wave32 waves
  mtrnn_fused<<<grid, block, 0, stream>>>(x, io_state, cf_state,
                                          Wi2io, bi2io, Wio2o, bio2o,
                                          Wio2io, bio2io, Wcf2io, bcf2io, out);
}
